// Dsa_Encoder_43722767073502
// MI455X (gfx1250) — matrix operand order repaired and checked by reading, not compiled, not run on hardware
//
#include <hip/hip_runtime.h>
#include <hip/hip_bf16.h>

// Problem dims (fixed by the reference)
#define TT 512   // time steps
#define HH 128   // hidden
#define LL 64    // latent
#define BB 8     // batch

typedef __attribute__((ext_vector_type(16))) _Float16 v16h;
typedef __attribute__((ext_vector_type(8)))  _Float16 v8h;
typedef __attribute__((ext_vector_type(8)))  float    v8f;

// ---------------------------------------------------------------------------
// Packed-f16 WMMA fragment loaders (CDNA5 16x16x32 f16, wave32, ISA 7.12.2).
// Memory must be K-contiguous; every fragment is exactly two 16-byte loads.
//
// A tile 16(M)x32(K): lane holds row m = lane&15; element i holds
//   k = (i&7) + 16*(i>>3) + 8*(lane>>4)
//   -> halves [kb..kb+7] and [kb+16..kb+23], kb = 8*(lane>>4)
// B tile 32(K)x16(N): lane holds col n = lane&15; element i holds
//   k = (i&7) + 16*(i>>3) + 8*(lane>>4)  (same as A)
// lda (in halves) must be a multiple of 8 for 16B alignment (64/256/512 all ok).
// ---------------------------------------------------------------------------
__device__ __forceinline__ v16h load_frag_a16(const _Float16* __restrict__ p,
                                              int lda, int lane) {
  const int m  = lane & 15;
  const int kb = (lane >> 4) << 3;
  const _Float16* r0 = p + m * lda + kb;
  const v8h lo = *(const v8h*)r0;
  const v8h hi = *(const v8h*)(r0 + 16);
  v16h r;
#pragma unroll
  for (int i = 0; i < 8; ++i) { r[i] = lo[i]; r[i + 8] = hi[i]; }
  return r;
}

__device__ __forceinline__ v16h load_frag_b16(const _Float16* __restrict__ p,
                                              int lda, int lane) {
  const int n  = lane & 15;
  const int kb = (lane >> 4) << 3;
  const _Float16* r0 = p + n * lda + kb;
  const v8h lo = *(const v8h*)r0;
  const v8h hi = *(const v8h*)(r0 + 16);
  v16h r;
#pragma unroll
  for (int i = 0; i < 8; ++i) { r[i] = lo[i]; r[i + 8] = hi[i]; }
  return r;
}

__device__ __forceinline__ float sigmoid_f(float x) {
  return 1.0f / (1.0f + __expf(-x));
}

// ---------------------------------------------------------------------------
// kc_cvt: plain f32 -> f16 conversion, 4 elements/thread (n multiple of 4).
// ---------------------------------------------------------------------------
__global__ __launch_bounds__(256) void kc_cvt(const float* __restrict__ in,
                                              _Float16* __restrict__ out,
                                              int n) {
  const int i = (blockIdx.x * 256 + threadIdx.x) * 4;
  if (i < n) {
    const float4 v = *(const float4*)(in + i);
    v8h dummy; (void)dummy;
    out[i + 0] = (_Float16)v.x;
    out[i + 1] = (_Float16)v.y;
    out[i + 2] = (_Float16)v.z;
    out[i + 3] = (_Float16)v.w;
  }
}

// ---------------------------------------------------------------------------
// kc_xt: xt16[b,t,l] = (f16) x[b,l,t]  — LDS tiled transpose, coalesced both ways.
// grid (T/16, L/16, B), block (16,16)
// ---------------------------------------------------------------------------
__global__ __launch_bounds__(256) void kc_xt(const float* __restrict__ x,
                                             _Float16* __restrict__ xt16) {
  __shared__ float tile[16][17];
  const int b  = blockIdx.z;
  const int t0 = blockIdx.x * 16;
  const int l0 = blockIdx.y * 16;
  const int tx = threadIdx.x, ty = threadIdx.y;
  tile[ty][tx] = x[(b * LL + l0 + ty) * TT + t0 + tx];   // coalesced over t
  __syncthreads();
  xt16[(b * TT + t0 + ty) * LL + l0 + tx] = (_Float16)tile[tx][ty]; // coalesced over l
}

// ---------------------------------------------------------------------------
// k0: hc[b,g] = b_ih[g] + b_hh[g] + sum_h h0[b,h]*W_hh[g,h]     (B x 4H)
// ---------------------------------------------------------------------------
__global__ __launch_bounds__(512) void k0_hc(const float* __restrict__ h0,
                                             const float* __restrict__ W_hh,
                                             const float* __restrict__ b_ih,
                                             const float* __restrict__ b_hh,
                                             float* __restrict__ hc) {
  const int g = threadIdx.x;
  const int b = blockIdx.x;
  float acc = b_ih[g] + b_hh[g];
  const float* h = h0 + b * HH;
  const float* w = W_hh + g * HH;
#pragma unroll 4
  for (int k = 0; k < HH; ++k) acc += h[k] * w[k];
  hc[b * (4 * HH) + g] = acc;
}

// ---------------------------------------------------------------------------
// k1: gates GEMM (16 t-rows x 512 gates, K=64) fused with LSTM cell.
//   gates[b,t,g] = sum_l xt16[b,t,l] * W_ih[g,l] + hc[b,g]
// 8 waves; wave w owns gate columns [64w, 64w+64).  Gates staged in LDS,
// then c = sig(f)*c0 + sig(i)*tanh(g); h = sig(o)*tanh(c).
// Writes h_all (f32 output) and q16 = [h | c] (f16 scratch, K-contiguous).
// ---------------------------------------------------------------------------
__global__ __launch_bounds__(256) void k1_gates_lstm(
    const _Float16* __restrict__ xt16, const _Float16* __restrict__ Wih16,
    const float* __restrict__ c0, const float* __restrict__ hc,
    float* __restrict__ h_all, _Float16* __restrict__ q16) {
  __shared__ float lds_g[16][4 * HH];                 // 32 KB
  const int b    = blockIdx.y;
  const int t0   = blockIdx.x * 16;
  const int lane = threadIdx.x & 31;
  const int wave = threadIdx.x >> 5;

  const _Float16* Ap = xt16 + (b * TT + t0) * LL;     // A(m,k): m*64 + k
  const v16h a0 = load_frag_a16(Ap,      LL, lane);
  const v16h a1 = load_frag_a16(Ap + 32, LL, lane);

  const int g0 = wave * 64;
#pragma unroll
  for (int j = 0; j < 4; ++j) {
    const int gn = g0 + 16 * j;
    const _Float16* Bp = Wih16 + gn * LL;             // B(k,n): n*64 + k
    const v16h b0 = load_frag_b16(Bp,      LL, lane);
    const v16h b1 = load_frag_b16(Bp + 32, LL, lane);
    v8f c = {};
    c = __builtin_amdgcn_wmma_f32_16x16x32_f16(false, a0, false, b0,
                                               (short)0, c, false, false);
    c = __builtin_amdgcn_wmma_f32_16x16x32_f16(false, a1, false, b1,
                                               (short)0, c, false, false);
    const int n  = lane & 15;
    const int mb = (lane >> 4) << 3;
    const float bias = hc[b * (4 * HH) + gn + n];
#pragma unroll
    for (int r = 0; r < 8; ++r) lds_g[mb + r][gn + n] = c[r] + bias;
  }
  __syncthreads();

#pragma unroll
  for (int e = threadIdx.x; e < 16 * HH; e += 256) {
    const int tm = e >> 7;
    const int h  = e & (HH - 1);
    const float iv = sigmoid_f(lds_g[tm][h]);
    const float fv = sigmoid_f(lds_g[tm][HH + h]);
    const float gv = tanhf(lds_g[tm][2 * HH + h]);
    const float ov = sigmoid_f(lds_g[tm][3 * HH + h]);
    const float cv = fv * c0[b * HH + h] + iv * gv;
    const float hv = ov * tanhf(cv);
    const int t = t0 + tm;
    h_all[(b * TT + t) * HH + h] = hv;
    q16[(b * TT + t) * (2 * HH) + h]      = (_Float16)hv;
    q16[(b * TT + t) * (2 * HH) + HH + h] = (_Float16)cv;
  }
}

// ---------------------------------------------------------------------------
// k2: w2q[b,t,s] = sum_h q[b,t,h]*W2[s,h] + b2[s]   (M=T, N=T, K=2H=256)
// grid (T/16, T/256, B), 4 waves; wave computes a 16x64 strip (4 C tiles),
// A fragment loaded once per K-chunk and reused across the strip.
// ---------------------------------------------------------------------------
__global__ __launch_bounds__(128) void k2_w2q(
    const _Float16* __restrict__ q16, const _Float16* __restrict__ W2h,
    const float* __restrict__ b2, float* __restrict__ w2q) {
  const int b    = blockIdx.z;
  const int t0   = blockIdx.x * 16;
  const int lane = threadIdx.x & 31;
  const int wave = threadIdx.x >> 5;
  const int n0   = blockIdx.y * 256 + wave * 64;
  const _Float16* Ap = q16 + (b * TT + t0) * (2 * HH);
  v8f c[4] = {};
#pragma unroll
  for (int kc = 0; kc < (2 * HH) / 32; ++kc) {
    const v16h af = load_frag_a16(Ap + kc * 32, 2 * HH, lane);
#pragma unroll
    for (int j = 0; j < 4; ++j) {
      const v16h bf =
          load_frag_b16(W2h + (n0 + 16 * j) * (2 * HH) + kc * 32, 2 * HH, lane);
      c[j] = __builtin_amdgcn_wmma_f32_16x16x32_f16(false, af, false, bf,
                                                    (short)0, c[j], false, false);
    }
  }
  const int n  = lane & 15;
  const int mb = (lane >> 4) << 3;
#pragma unroll
  for (int j = 0; j < 4; ++j) {
    const float bias = b2[n0 + 16 * j + n];
#pragma unroll
    for (int r = 0; r < 8; ++r)
      w2q[(b * TT + t0 + mb + r) * TT + n0 + 16 * j + n] = c[j][r] + bias;
  }
}

// ---------------------------------------------------------------------------
// k3: w1x[b,l,t'] = sum_s x[b,l,s]*W1[t',s] + b1[t']  (M=L=64, N=T, K=T=512)
// Same 16x64-strip-per-wave structure as k2.
// ---------------------------------------------------------------------------
__global__ __launch_bounds__(128) void k3_w1x(
    const _Float16* __restrict__ x16, const _Float16* __restrict__ W1h,
    const float* __restrict__ b1, float* __restrict__ w1x) {
  const int b    = blockIdx.z;
  const int l0   = blockIdx.x * 16;
  const int lane = threadIdx.x & 31;
  const int wave = threadIdx.x >> 5;
  const int n0   = blockIdx.y * 256 + wave * 64;
  const _Float16* Ap = x16 + (b * LL + l0) * TT;
  v8f c[4] = {};
  for (int kc = 0; kc < TT / 32; ++kc) {
    const v16h af = load_frag_a16(Ap + kc * 32, TT, lane);
#pragma unroll
    for (int j = 0; j < 4; ++j) {
      const v16h bf = load_frag_b16(W1h + (n0 + 16 * j) * TT + kc * 32, TT, lane);
      c[j] = __builtin_amdgcn_wmma_f32_16x16x32_f16(false, af, false, bf,
                                                    (short)0, c[j], false, false);
    }
  }
  const int n  = lane & 15;
  const int mb = (lane >> 4) << 3;
#pragma unroll
  for (int j = 0; j < 4; ++j) {
    const float bias = b1[n0 + 16 * j + n];
#pragma unroll
    for (int r = 0; r < 8; ++r)
      w1x[(b * LL + l0 + mb + r) * TT + n0 + 16 * j + n] = c[j][r] + bias;
  }
}

// ---------------------------------------------------------------------------
// k4: one block per (b,t).
//   s[l]   = sum_s tanh(w1x[b,l,s] + w2q[b,t,s]) * v_w[s] + v_b
//   alpha  = log_softmax_l(s);  out[b,t,l] = x[b,l,t] * alpha[l]
// 8 waves; each wave owns 8 of the 64 l-rows; wave32 shuffle reductions.
// ---------------------------------------------------------------------------
__global__ __launch_bounds__(256) void k4_attn(
    const float* __restrict__ x, const float* __restrict__ w1x,
    const float* __restrict__ w2q, const float* __restrict__ v_w,
    const float* __restrict__ v_b, float* __restrict__ out) {
  __shared__ float w2q_sh[TT];
  __shared__ float vw_sh[TT];
  __shared__ float s_sh[LL];
  __shared__ float lse_sh;
  const int b   = blockIdx.y;
  const int t   = blockIdx.x;
  const int tid = threadIdx.x;
  const float* wq = w2q + (b * TT + t) * TT;
  for (int i = tid; i < TT; i += 256) {
    w2q_sh[i] = wq[i];
    vw_sh[i]  = v_w[i];
  }
  __syncthreads();

  const int wave = tid >> 5, lane = tid & 31;
  const float vb = v_b[0];
#pragma unroll
  for (int j = 0; j < LL / 8; ++j) {
    const int l = wave * (LL / 8) + j;
    const float* wx = w1x + (b * LL + l) * TT;
    float acc = 0.0f;
    for (int s = lane; s < TT; s += 32)
      acc += tanhf(wx[s] + w2q_sh[s]) * vw_sh[s];
#pragma unroll
    for (int off = 16; off > 0; off >>= 1)
      acc += __shfl_xor(acc, off, 32);
    if (lane == 0) s_sh[l] = acc + vb;
  }
  __syncthreads();

  if (tid < 32) {
    const float a0 = s_sh[tid], a1 = s_sh[tid + 32];
    float mx = fmaxf(a0, a1);
#pragma unroll
    for (int off = 16; off > 0; off >>= 1)
      mx = fmaxf(mx, __shfl_xor(mx, off, 32));
    float e = expf(a0 - mx) + expf(a1 - mx);
#pragma unroll
    for (int off = 16; off > 0; off >>= 1)
      e += __shfl_xor(e, off, 32);
    if (tid == 0) lse_sh = mx + logf(e);
  }
  __syncthreads();

  if (tid < LL) {
    const float alpha = s_sh[tid] - lse_sh;
    out[(b * TT + t) * LL + tid] = x[(b * LL + tid) * TT + t] * alpha;
  }
}

// ---------------------------------------------------------------------------
// Host launcher.  Inputs (all f32, setup_inputs order):
//  0:x 1:h0 2:c0 3:W_ih 4:W_hh 5:b_ih 6:b_hh 7:W1 8:b1 9:W2 10:b2 11:v_w 12:v_b
// d_out = [ output (B,T,L) | h_all (B,T,H) ]  f32
// Workspace layout (bytes):
//   w2q f32  8 MB | w1x f32 1 MB | hc f32 16 KB |
//   x16 512 KB | xt16 512 KB | Wih16 64 KB | W1h 512 KB | W2h 256 KB | q16 2 MB
// ---------------------------------------------------------------------------
extern "C" void kernel_launch(void* const* d_in, const int* in_sizes, int n_in,
                              void* d_out, int out_size, void* d_ws,
                              size_t ws_size, hipStream_t stream) {
  (void)in_sizes; (void)n_in; (void)out_size; (void)ws_size;
  const float* x    = (const float*)d_in[0];
  const float* h0   = (const float*)d_in[1];
  const float* c0   = (const float*)d_in[2];
  const float* W_ih = (const float*)d_in[3];
  const float* W_hh = (const float*)d_in[4];
  const float* b_ih = (const float*)d_in[5];
  const float* b_hh = (const float*)d_in[6];
  const float* W1   = (const float*)d_in[7];
  const float* b1   = (const float*)d_in[8];
  const float* W2   = (const float*)d_in[9];
  const float* b2   = (const float*)d_in[10];
  const float* v_w  = (const float*)d_in[11];
  const float* v_b  = (const float*)d_in[12];

  float* out   = (float*)d_out;
  float* h_all = out + BB * TT * LL;

  char* w = (char*)d_ws;
  float*    w2q   = (float*)w;              w += sizeof(float) * BB * TT * TT;
  float*    w1x   = (float*)w;              w += sizeof(float) * BB * LL * TT;
  float*    hc    = (float*)w;              w += sizeof(float) * BB * 4 * HH;
  _Float16* x16   = (_Float16*)w;           w += sizeof(_Float16) * BB * LL * TT;
  _Float16* xt16  = (_Float16*)w;           w += sizeof(_Float16) * BB * TT * LL;
  _Float16* Wih16 = (_Float16*)w;           w += sizeof(_Float16) * 4 * HH * LL;
  _Float16* W1h   = (_Float16*)w;           w += sizeof(_Float16) * TT * TT;
  _Float16* W2h   = (_Float16*)w;           w += sizeof(_Float16) * TT * 2 * HH;
  _Float16* q16   = (_Float16*)w;           // BB*TT*2*HH

  // --- one-shot f32->f16 prep (K-contiguous operand layouts) ---
  hipLaunchKernelGGL(kc_cvt, dim3((BB * LL * TT) / 1024), dim3(256), 0, stream,
                     x, x16, BB * LL * TT);
  hipLaunchKernelGGL(kc_cvt, dim3((4 * HH * LL) / 1024), dim3(256), 0, stream,
                     W_ih, Wih16, 4 * HH * LL);
  hipLaunchKernelGGL(kc_cvt, dim3((TT * TT) / 1024), dim3(256), 0, stream,
                     W1, W1h, TT * TT);
  hipLaunchKernelGGL(kc_cvt, dim3((TT * 2 * HH) / 1024), dim3(256), 0, stream,
                     W2, W2h, TT * 2 * HH);
  hipLaunchKernelGGL(kc_xt, dim3(TT / 16, LL / 16, BB), dim3(16, 16), 0, stream,
                     x, xt16);
  hipLaunchKernelGGL(k0_hc, dim3(BB), dim3(4 * HH), 0, stream,
                     h0, W_hh, b_ih, b_hh, hc);

  // --- WMMA pipeline ---
  hipLaunchKernelGGL(k1_gates_lstm, dim3(TT / 16, BB), dim3(256), 0, stream,
                     xt16, Wih16, c0, hc, h_all, q16);
  hipLaunchKernelGGL(k2_w2q, dim3(TT / 16, TT / 256, BB), dim3(128), 0, stream,
                     q16, W2h, b2, w2q);
  hipLaunchKernelGGL(k3_w1x, dim3(LL / 16, TT / 256, BB), dim3(128), 0, stream,
                     x16, W1h, b1, w1x);
  hipLaunchKernelGGL(k4_attn, dim3(TT, BB), dim3(256), 0, stream,
                     x, w1x, w2q, v_w, v_b, out);
}
